// KPTransformerResidualBlock_47957604827529
// MI455X (gfx1250) — matrix operand order repaired and checked by reading, not compiled, not run on hardware
//
#include <hip/hip_runtime.h>
#include <math.h>

#define NPTS   50000
#define HNBR   32
#define CIN    512
#define CMID   128
#define KP     15
#define CPG    16
#define KCPG   (KP * CPG)            // 240
#define MPAD   50176                 // 392*128, padded row count for workspace
#define MT     ((NPTS + 127) / 128)  // 391 row-blocks

#ifndef __has_builtin
#define __has_builtin(x) 0
#endif
#if __has_builtin(__builtin_amdgcn_tensor_load_to_lds)
#define HAVE_TDM 1
#else
#define HAVE_TDM 0
#endif

typedef __attribute__((ext_vector_type(16))) __bf16 v16bf;
typedef __attribute__((ext_vector_type(8)))  float  v8f;
typedef __attribute__((ext_vector_type(2)))  __bf16 bf16x2;
typedef __attribute__((ext_vector_type(4)))  unsigned int v4u;
typedef __attribute__((ext_vector_type(8)))  int v8i;
typedef __attribute__((ext_vector_type(4)))  int v4i;

union U32BF2 { unsigned int u; bf16x2 h; };

__device__ __forceinline__ unsigned short f2bf(float f) {
  union { float f; unsigned int u; } v; v.f = f;
  unsigned int r = v.u + 0x7FFFu + ((v.u >> 16) & 1u);   // round-to-nearest-even
  return (unsigned short)(r >> 16);
}

__device__ __forceinline__ float leaky(float x) { return fmaxf(x, 0.1f * x); }

// A fragment: 16x32 bf16, per ISA layout (lanes 0-15: M=lane, K pairs 0..7/16..23;
// lanes 16-31: same rows, K offset +8).
__device__ __forceinline__ v16bf load_a_frag(const unsigned short* lds, int stride,
                                             int mbase, int kbase, int lane) {
  const int m = mbase + (lane & 15);
  const int half = lane >> 4;
  v16bf a;
#pragma unroll
  for (int j = 0; j < 8; ++j) {
    const int kb = kbase + 2 * j + ((j >> 2) << 3) + (half << 3);
    U32BF2 cv; cv.u = *(const unsigned int*)(lds + m * stride + kb);
    a[2 * j] = cv.h[0]; a[2 * j + 1] = cv.h[1];
  }
  return a;
}

// B fragment: 32x16 bf16 from n-major LDS (BT[n][k]); lanes 0-15: N=lane, K 0..7/16..23;
// lanes 16-31: N=lane-16, K offset +8.
__device__ __forceinline__ v16bf load_b_frag(const unsigned short* lds, int stride,
                                             int nbase, int kbase, int lane) {
  const int n = nbase + (lane & 15);
  const int koff = kbase + ((lane >> 4) << 3);
  v16bf b;
#pragma unroll
  for (int j = 0; j < 8; ++j) {
    U32BF2 cv; cv.u = *(const unsigned int*)(lds + n * stride + koff + 2 * j + ((j >> 2) << 3));
    b[2 * j] = cv.h[0]; b[2 * j + 1] = cv.h[1];
  }
  return b;
}

__device__ __forceinline__ v8f wmma_bf16(v16bf a, v16bf b, v8f c) {
  return __builtin_amdgcn_wmma_f32_16x16x32_bf16(false, a, false, b, (short)0, c, false, false);
}

__device__ __forceinline__ void wait_tensorcnt0() {
#if __has_builtin(__builtin_amdgcn_s_wait_tensorcnt)
  __builtin_amdgcn_s_wait_tensorcnt(0);
#else
  asm volatile("s_wait_tensorcnt 0x0" ::: "memory");
#endif
}

#if HAVE_TDM
// Issue one 2D TDM tile load: global (row-major, elements = bf16) -> LDS.
// tile_d0 elements per row (dim0, contiguous), tile_d1 rows; tensor dims give
// OOB-zero bounds; stride_d0 = global row stride in elements.
// LDS rows padded: pad_interval_code (1<<c+1 dwords), pad_amount_code (c+1 dwords).
__device__ __forceinline__ void tdm_load_2d_bf16(unsigned lds_addr, const void* gptr,
                                                 unsigned tile_d0, unsigned tile_d1,
                                                 unsigned tensor_d0, unsigned tensor_d1,
                                                 unsigned stride_d0,
                                                 unsigned pad_interval_code,
                                                 unsigned pad_amount_code) {
  const unsigned long long ga = (unsigned long long)gptr;
  v4u g0;
  g0[0] = 1u;                                              // count=1, user descriptor
  g0[1] = lds_addr;                                        // LDS byte address
  g0[2] = (unsigned)(ga & 0xFFFFFFFFu);                    // global_addr[31:0]
  g0[3] = (unsigned)((ga >> 32) & 0x01FFFFFFu) | (2u << 30); // addr[56:32] | type=2
  v8i g1;
  g1[0] = (int)((1u << 16) |                               // data_size = 2B
                (1u << 20) |                               // pad_enable
                (pad_interval_code << 22) |
                (pad_amount_code << 25));
  g1[1] = (int)((tensor_d0 & 0xFFFFu) << 16);              // tensor_dim0[15:0]
  g1[2] = (int)(((tensor_d0 >> 16) & 0xFFFFu) | ((tensor_d1 & 0xFFFFu) << 16));
  g1[3] = (int)(((tensor_d1 >> 16) & 0xFFFFu) | ((tile_d0 & 0xFFFFu) << 16));
  g1[4] = (int)(tile_d1 & 0xFFFFu);                        // tile_dim1 (tile_dim2=0)
  g1[5] = (int)stride_d0;                                  // tensor_dim0_stride[31:0]
  g1[6] = 0;
  g1[7] = 0;
  v4i z4; z4[0] = 0; z4[1] = 0; z4[2] = 0; z4[3] = 0;      // groups 2/3 unused (2D)
#if __clang_major__ >= 23
  v8i z8; z8[0] = 0; z8[1] = 0; z8[2] = 0; z8[3] = 0;
  z8[4] = 0; z8[5] = 0; z8[6] = 0; z8[7] = 0;
  __builtin_amdgcn_tensor_load_to_lds(g0, g1, z4, z4, z8, 0);
#else
  __builtin_amdgcn_tensor_load_to_lds(g0, g1, z4, z4, 0);
#endif
}
#endif  // HAVE_TDM

// ---------------- weight transpose + fp32->bf16 convert ----------------
// src: R x C row-major fp32; dst: C x R row-major bf16 (dst[c*R + r] = src[r*C + c])
__global__ __launch_bounds__(256)
void k_transpose_bf16(const float* __restrict__ src, unsigned short* __restrict__ dst,
                      int R, int C) {
  const int i = blockIdx.x * 256 + threadIdx.x;
  if (i < R * C) {
    const int r = i / C, c = i % C;
    dst[c * R + r] = f2bf(src[i]);
  }
}

// ---------------- kernel 1: X = leaky(s_feats @ W_down + b_down) ----------------
__global__ __launch_bounds__(256)
void k_down_gemm(const float* __restrict__ s_feats, const unsigned short* __restrict__ WdT,
                 const float* __restrict__ b_down, float* __restrict__ X,
                 unsigned short* __restrict__ Xbf) {
  __shared__ unsigned short sA[128 * 34];   // stride 34 (17 dwords, conflict-free)
  __shared__ unsigned short sBT[128 * 34];
  const int tid = threadIdx.x, lane = tid & 31, wave = tid >> 5;
  const int m0 = blockIdx.x * 128;
  v8f acc[8];
#pragma unroll
  for (int t = 0; t < 8; ++t)
#pragma unroll
    for (int e = 0; e < 8; ++e) acc[t][e] = 0.f;

  for (int kc = 0; kc < CIN; kc += 32) {
    __syncthreads();
    // stage A: 128 rows x 32 k fp32 -> bf16 (4 float4 per thread)
#pragma unroll
    for (int i = 0; i < 4; ++i) {
      const int li = tid + i * 256;        // 0..1023
      const int r = li >> 3;
      const int kk = (li & 7) << 2;
      const int grow = m0 + r;
      float4 v = make_float4(0.f, 0.f, 0.f, 0.f);
      if (grow < NPTS) v = *(const float4*)(s_feats + (size_t)grow * CIN + kc + kk);
      unsigned short* p = sA + r * 34 + kk;
      *(unsigned int*)(p)     = (unsigned)f2bf(v.x) | ((unsigned)f2bf(v.y) << 16);
      *(unsigned int*)(p + 2) = (unsigned)f2bf(v.z) | ((unsigned)f2bf(v.w) << 16);
    }
    // stage BT: 128 n x 32 k (bf16 pairs, contiguous in WdT[n][k])
#pragma unroll
    for (int i = 0; i < 8; ++i) {
      const int li = tid + i * 256;        // 0..2047
      const int n = li >> 4;
      const int kk = (li & 15) << 1;
      const unsigned int u = *(const unsigned int*)(WdT + (size_t)n * CIN + kc + kk);
      *(unsigned int*)(sBT + n * 34 + kk) = u;
    }
    __syncthreads();
    const v16bf af = load_a_frag(sA, 34, wave * 16, 0, lane);
#pragma unroll
    for (int t = 0; t < 8; ++t)
      acc[t] = wmma_bf16(af, load_b_frag(sBT, 34, t * 16, 0, lane), acc[t]);
  }
  const int n = lane & 15, half = lane >> 4;
  const int rb = m0 + wave * 16 + half * 8;
#pragma unroll
  for (int t = 0; t < 8; ++t) {
    const int col = t * 16 + n;
    const float bias = b_down[col];
#pragma unroll
    for (int r = 0; r < 8; ++r) {
      const int row = rb + r;
      if (row < NPTS) {
        const float v = leaky(acc[t][r] + bias);
        X[(size_t)row * CMID + col] = v;
        Xbf[(size_t)row * CMID + col] = f2bf(v);
      }
    }
  }
}

// ---------------- kernel 2a: H1 = leaky(X @ W_mlp1 + b_mlp1) (bf16 out) ----------------
__global__ __launch_bounds__(256)
void k_mlp1_gemm(const unsigned short* __restrict__ Xbf, const unsigned short* __restrict__ Wm1T,
                 const float* __restrict__ b1, unsigned short* __restrict__ H1bf) {
  __shared__ unsigned short sA[128 * 34];
  __shared__ unsigned short sBT[128 * 34];
  const int tid = threadIdx.x, lane = tid & 31, wave = tid >> 5;
  const int m0 = blockIdx.x * 128;
  v8f acc[8];
#pragma unroll
  for (int t = 0; t < 8; ++t)
#pragma unroll
    for (int e = 0; e < 8; ++e) acc[t][e] = 0.f;

  for (int kc = 0; kc < CMID; kc += 32) {
    __syncthreads();
#pragma unroll
    for (int i = 0; i < 8; ++i) {
      const int li = tid + i * 256;        // 0..2047
      const int r = li >> 4;
      const int kk = (li & 15) << 1;
      const int grow = m0 + r;
      unsigned int u = 0;
      if (grow < NPTS) u = *(const unsigned int*)(Xbf + (size_t)grow * CMID + kc + kk);
      *(unsigned int*)(sA + r * 34 + kk) = u;
    }
#pragma unroll
    for (int i = 0; i < 8; ++i) {
      const int li = tid + i * 256;
      const int n = li >> 4;
      const int kk = (li & 15) << 1;
      const unsigned int u = *(const unsigned int*)(Wm1T + (size_t)n * CMID + kc + kk);
      *(unsigned int*)(sBT + n * 34 + kk) = u;
    }
    __syncthreads();
    const v16bf af = load_a_frag(sA, 34, wave * 16, 0, lane);
#pragma unroll
    for (int t = 0; t < 8; ++t)
      acc[t] = wmma_bf16(af, load_b_frag(sBT, 34, t * 16, 0, lane), acc[t]);
  }
  const int n = lane & 15, half = lane >> 4;
  const int rb = m0 + wave * 16 + half * 8;
#pragma unroll
  for (int t = 0; t < 8; ++t) {
    const int col = t * 16 + n;
    const float bias = b1[col];
#pragma unroll
    for (int r = 0; r < 8; ++r) {
      const int row = rb + r;
      if (row < NPTS) H1bf[(size_t)row * CMID + col] = f2bf(leaky(acc[t][r] + bias));
    }
  }
}

// ---------------- kernel 2b: SIG = sigmoid(H1 @ W_mlp2), (N x 240) fp32 ----------------
__global__ __launch_bounds__(256)
void k_mlp2_gemm(const unsigned short* __restrict__ H1bf, const unsigned short* __restrict__ Wm2T,
                 float* __restrict__ SIG) {
  __shared__ unsigned short sA[128 * 34];
  __shared__ unsigned short sBT[240 * 34];
  const int tid = threadIdx.x, lane = tid & 31, wave = tid >> 5;
  const int m0 = blockIdx.x * 128;
  v8f acc[15];
#pragma unroll
  for (int t = 0; t < 15; ++t)
#pragma unroll
    for (int e = 0; e < 8; ++e) acc[t][e] = 0.f;

  for (int kc = 0; kc < CMID; kc += 32) {
    __syncthreads();
#pragma unroll
    for (int i = 0; i < 8; ++i) {
      const int li = tid + i * 256;
      const int r = li >> 4;
      const int kk = (li & 15) << 1;
      const int grow = m0 + r;
      unsigned int u = 0;
      if (grow < NPTS) u = *(const unsigned int*)(H1bf + (size_t)grow * CMID + kc + kk);
      *(unsigned int*)(sA + r * 34 + kk) = u;
    }
#pragma unroll
    for (int i = 0; i < 15; ++i) {          // 240*16 = 3840 uints, 15 per thread
      const int li = tid + i * 256;
      const int n = li >> 4;
      const int kk = (li & 15) << 1;
      const unsigned int u = *(const unsigned int*)(Wm2T + (size_t)n * CMID + kc + kk);
      *(unsigned int*)(sBT + n * 34 + kk) = u;
    }
    __syncthreads();
    const v16bf af = load_a_frag(sA, 34, wave * 16, 0, lane);
#pragma unroll
    for (int t = 0; t < 15; ++t)
      acc[t] = wmma_bf16(af, load_b_frag(sBT, 34, t * 16, 0, lane), acc[t]);
  }
  const int n = lane & 15, half = lane >> 4;
  const int rb = m0 + wave * 16 + half * 8;
#pragma unroll
  for (int t = 0; t < 15; ++t) {
    const int col = t * 16 + n;
#pragma unroll
    for (int r = 0; r < 8; ++r) {
      const int row = rb + r;
      if (row < NPTS)
        SIG[(size_t)row * KCPG + col] = 1.0f / (1.0f + __expf(-acc[t][r]));
    }
  }
}

// ---------------- kernel 3: geometry argmin + gather-aggregate (wave per point) ----------------
__global__ __launch_bounds__(256)
void k_aggregate(const float* __restrict__ q_pts, const float* __restrict__ s_pts,
                 const int* __restrict__ inds, const float* __restrict__ kpts_g,
                 const float* __restrict__ kpw_g, const float* __restrict__ SIG,
                 const float* __restrict__ X, unsigned short* __restrict__ Ybf) {
  __shared__ float kpts[KP * 3];
  __shared__ float kpw[KP * CMID];
  __shared__ float sig[8][KCPG];
  const int tid = threadIdx.x, lane = tid & 31, wave = tid >> 5;
  for (int i = tid; i < KP * 3; i += 256) kpts[i] = kpts_g[i];
  for (int i = tid; i < KP * CMID; i += 256) kpw[i] = kpw_g[i];
  const int m = blockIdx.x * 8 + wave;
  if (m < NPTS)
    for (int i = lane; i < KCPG; i += 32) sig[wave][i] = SIG[(size_t)m * KCPG + i];
  __syncthreads();
  if (m >= NPTS) return;

  const float qx = q_pts[m * 3 + 0], qy = q_pts[m * 3 + 1], qz = q_pts[m * 3 + 2];
  const int idx = inds[(size_t)m * HNBR + lane];
  const float px = s_pts[idx * 3 + 0] - qx;
  const float py = s_pts[idx * 3 + 1] - qy;
  const float pz = s_pts[idx * 3 + 2] - qz;
  float best = 3.4e38f; int bk = 0;
#pragma unroll
  for (int k = 0; k < KP; ++k) {
    const float dx = px - kpts[k * 3 + 0];
    const float dy = py - kpts[k * 3 + 1];
    const float dz = pz - kpts[k * 3 + 2];
    const float d2 = dx * dx + dy * dy + dz * dz;
    if (d2 < best) { best = d2; bk = k; }
  }
  const float infl = fmaxf(0.f, 1.f - sqrtf(best) * (1.0f / 0.7f));

  const int c0 = lane * 4;     // 4 channels per lane; c>>3 == lane>>1 for all 4
  float a0 = 0.f, a1 = 0.f, a2 = 0.f, a3 = 0.f;
  for (int hh = 0; hh < HNBR; ++hh) {
    const int   bidx = __shfl(idx, hh, 32);
    const int   kk   = __shfl(bk, hh, 32);
    const float w    = __shfl(infl, hh, 32);
    const float sw   = sig[wave][kk * CPG + (lane >> 1)] * w;
    const float4 xv = *(const float4*)(X + (size_t)bidx * CMID + c0);
    const float4 kw = *(const float4*)(&kpw[kk * CMID + c0]);
    a0 += xv.x * kw.x * sw;
    a1 += xv.y * kw.y * sw;
    a2 += xv.z * kw.z * sw;
    a3 += xv.w * kw.w * sw;
  }
  unsigned int* dst = (unsigned int*)(Ybf + (size_t)m * CMID + c0);
  dst[0] = (unsigned)f2bf(a0) | ((unsigned)f2bf(a1) << 16);
  dst[1] = (unsigned)f2bf(a2) | ((unsigned)f2bf(a3) << 16);
}

// ---------------- kernel 4: out = leaky(Y @ W_up + b_up + s_feats) ----------------
// A/B tiles staged by the Tensor Data Mover (TDM) when available:
// row stride 128 bf16 = 64 dwords, pad 3 dwords -> LDS stride 67 dwords = 134 bf16
// (gcd(67,64)=1: conflict-free fragment loads). OOB rows (>= NPTS) read as zero
// via tensor_dim1, replacing the manual row guard.
#define UPSTRIDE 134
__global__ __launch_bounds__(256)
void k_up_gemm(const unsigned short* __restrict__ Ybf, const unsigned short* __restrict__ WupT,
               const float* __restrict__ b_up, const float* __restrict__ s_feats,
               float* __restrict__ out) {
  __shared__ unsigned short sA[128 * UPSTRIDE];
  __shared__ unsigned short sBT[128 * UPSTRIDE];
  const int tid = threadIdx.x, lane = tid & 31, wave = tid >> 5;
  const int m0 = blockIdx.x * 128;
  const int n0 = blockIdx.y * 128;

#if HAVE_TDM
  if (wave == 0) {
    // A tile: Ybf rows m0..m0+127 (OOB-zero past NPTS), 128 bf16/row.
    tdm_load_2d_bf16((unsigned)(unsigned long long)&sA[0],
                     Ybf + (size_t)m0 * CMID,
                     /*tile_d0=*/CMID, /*tile_d1=*/128,
                     /*tensor_d0=*/CMID, /*tensor_d1=*/(unsigned)(NPTS - m0),
                     /*stride_d0=*/CMID,
                     /*pad_interval=64dw*/5u, /*pad=3dw*/2u);
    // B tile: WupT rows n0..n0+127, 128 bf16/row (always in-bounds).
    tdm_load_2d_bf16((unsigned)(unsigned long long)&sBT[0],
                     WupT + (size_t)n0 * CMID,
                     CMID, 128, CMID, 128, CMID, 5u, 2u);
    wait_tensorcnt0();
  }
  __syncthreads();
#else
  // fallback: manual staging (same LDS layout)
#pragma unroll
  for (int i = 0; i < 32; ++i) {
    const int li = tid + i * 256;            // 0..8191
    const int r = li >> 6;
    const int kk = (li & 63) << 1;
    const int grow = m0 + r;
    unsigned int u = 0;
    if (grow < NPTS) u = *(const unsigned int*)(Ybf + (size_t)grow * CMID + kk);
    *(unsigned int*)(sA + r * UPSTRIDE + kk) = u;
  }
#pragma unroll
  for (int i = 0; i < 32; ++i) {
    const int li = tid + i * 256;
    const int nn = li >> 6;
    const int kk = (li & 63) << 1;
    const unsigned int u = *(const unsigned int*)(WupT + (size_t)(n0 + nn) * CMID + kk);
    *(unsigned int*)(sBT + nn * UPSTRIDE + kk) = u;
  }
  __syncthreads();
#endif

  v8f acc[8];
#pragma unroll
  for (int t = 0; t < 8; ++t)
#pragma unroll
    for (int e = 0; e < 8; ++e) acc[t][e] = 0.f;

#pragma unroll
  for (int kc = 0; kc < CMID; kc += 32) {
    const v16bf af = load_a_frag(sA, UPSTRIDE, wave * 16, kc, lane);
#pragma unroll
    for (int t = 0; t < 8; ++t)
      acc[t] = wmma_bf16(af, load_b_frag(sBT, UPSTRIDE, t * 16, kc, lane), acc[t]);
  }

  const int n = lane & 15, half = lane >> 4;
  const int rb = m0 + wave * 16 + half * 8;
#pragma unroll
  for (int t = 0; t < 8; ++t) {
    const int col = n0 + t * 16 + n;
    const float bias = b_up[col];
#pragma unroll
    for (int r = 0; r < 8; ++r) {
      const int row = rb + r;
      if (row < NPTS) {
        const float v = acc[t][r] + bias + s_feats[(size_t)row * CIN + col];
        out[(size_t)row * CIN + col] = leaky(v);
      }
    }
  }
}

// ---------------- host launcher ----------------
extern "C" void kernel_launch(void* const* d_in, const int* in_sizes, int n_in,
                              void* d_out, int out_size, void* d_ws, size_t ws_size,
                              hipStream_t stream) {
  (void)in_sizes; (void)n_in; (void)out_size; (void)ws_size;
  const float* q_pts   = (const float*)d_in[0];
  const float* s_pts   = (const float*)d_in[1];
  const float* s_feats = (const float*)d_in[2];
  const int*   inds    = (const int*)d_in[3];
  const float* kpts    = (const float*)d_in[4];
  const float* W_down  = (const float*)d_in[5];
  const float* b_down  = (const float*)d_in[6];
  const float* W_mlp1  = (const float*)d_in[7];
  const float* b_mlp1  = (const float*)d_in[8];
  const float* W_mlp2  = (const float*)d_in[9];
  const float* kp_w    = (const float*)d_in[10];
  const float* W_up    = (const float*)d_in[11];
  const float* b_up    = (const float*)d_in[12];
  float* out = (float*)d_out;

  char* ws = (char*)d_ws;
  size_t off = 0;
  auto take = [&](size_t bytes) -> char* {
    off = (off + 255) & ~(size_t)255;
    char* p = ws + off;
    off += bytes;
    return p;
  };
  float*          X    = (float*)take((size_t)MPAD * CMID * 4);
  unsigned short* Xbf  = (unsigned short*)take((size_t)MPAD * CMID * 2);
  unsigned short* H1bf = (unsigned short*)take((size_t)MPAD * CMID * 2);
  float*          SIG  = (float*)take((size_t)MPAD * KCPG * 4);
  unsigned short* Ybf  = (unsigned short*)take((size_t)MPAD * CMID * 2);
  unsigned short* WdT  = (unsigned short*)take((size_t)CIN * CMID * 2);
  unsigned short* Wm1T = (unsigned short*)take((size_t)CMID * CMID * 2);
  unsigned short* Wm2T = (unsigned short*)take((size_t)KCPG * CMID * 2);
  unsigned short* WupT = (unsigned short*)take((size_t)CIN * CMID * 2);

  // weight converts (transposed, bf16)
  k_transpose_bf16<<<(CIN * CMID + 255) / 256, 256, 0, stream>>>(W_down, WdT, CIN, CMID);
  k_transpose_bf16<<<(CMID * CMID + 255) / 256, 256, 0, stream>>>(W_mlp1, Wm1T, CMID, CMID);
  k_transpose_bf16<<<(CMID * KCPG + 255) / 256, 256, 0, stream>>>(W_mlp2, Wm2T, CMID, KCPG);
  k_transpose_bf16<<<(CMID * CIN + 255) / 256, 256, 0, stream>>>(W_up, WupT, CMID, CIN);

  k_down_gemm<<<MT, 256, 0, stream>>>(s_feats, WdT, b_down, X, Xbf);
  k_mlp1_gemm<<<MT, 256, 0, stream>>>(Xbf, Wm1T, b_mlp1, H1bf);
  k_mlp2_gemm<<<MT, 256, 0, stream>>>(H1bf, Wm2T, SIG);
  k_aggregate<<<(NPTS + 7) / 8, 256, 0, stream>>>(q_pts, s_pts, inds, kpts, kp_w, SIG, X, Ybf);
  k_up_gemm<<<dim3(MT, 4), 256, 0, stream>>>(Ybf, WupT, b_up, s_feats, out);
}
